// Transformer_16475494548202
// MI455X (gfx1250) — matrix operand order repaired and checked by reading, not compiled, not run on hardware
//
#include <hip/hip_runtime.h>
#include <hip/hip_bf16.h>
#include <math.h>

// ---------------------------------------------------------------------------
// Types
// ---------------------------------------------------------------------------
typedef __bf16 bf16;
typedef __attribute__((ext_vector_type(4)))  bf16  bf16x4;
typedef __attribute__((ext_vector_type(8)))  bf16  bf16x8;
typedef __attribute__((ext_vector_type(16))) bf16  bf16x16;
typedef __attribute__((ext_vector_type(8)))  float v8f;

// CDNA5 async global->LDS path (ASYNCcnt), guarded so absence can't break build.
// Probe round 3 revealed the builtin signature: (v4i32 AS1*, v4i32 AS3*, Ii, Ii).
#if __has_builtin(__builtin_amdgcn_global_load_async_to_lds_b128) && \
    __has_builtin(__builtin_amdgcn_s_wait_asynccnt)
#define HAVE_ASYNC_LDS 1
typedef __attribute__((ext_vector_type(4))) int i32x4;
typedef __attribute__((address_space(1))) i32x4 as1_i32x4;
typedef __attribute__((address_space(3))) i32x4 as3_i32x4;
#else
#define HAVE_ASYNC_LDS 0
#endif

__device__ __forceinline__ bf16x16 comb16(const bf16* lo, const bf16* hi) {
    bf16x8 a = *(const bf16x8*)lo;
    bf16x8 b = *(const bf16x8*)hi;
    return __builtin_shufflevector(a, b, 0,1,2,3,4,5,6,7,8,9,10,11,12,13,14,15);
}

// ---------------------------------------------------------------------------
// Generic GEMM:  C[M,N] = act( A[M,K] * W[N,K]^T + bias + res )
// bf16 WMMA 16x16x32, f32 accumulate. Tiles: 128x128x32, 256 thr (8 waves).
// Pipelined: global loads for tile kb+1 are issued BEFORE the WMMA block of
// tile kb; cvt+LDS-store happens after, so loads overlap matrix math.
// act: 0 = none, 1 = exact GELU, 2 = QKV-split epilogue (writes bf16 Q/K/V).
// ---------------------------------------------------------------------------
#define GBM 128
#define GBN 128
#define GBK 32
#define GLDK 40   // padded row stride (bf16 elems), keeps 16B alignment

__global__ __launch_bounds__(256) void gemm_bf16_kernel(
    const float* __restrict__ A, const float* __restrict__ W,
    const float* __restrict__ bias, const float* __restrict__ res,
    float* __restrict__ C, bf16* __restrict__ Qo, bf16* __restrict__ Ko,
    bf16* __restrict__ Vo, int M, int N, int K, int act)
{
    __shared__ __attribute__((aligned(16))) bf16 As[2][GBM][GLDK];
    __shared__ __attribute__((aligned(16))) bf16 Bs[2][GBN][GLDK];

    const int tid  = threadIdx.x;
    const int lane = tid & 31;
    const int wave = tid >> 5;
    const int wm   = wave >> 1;       // 0..3 : 32-row stripe
    const int wn   = wave & 1;        // 0..1 : 64-col stripe
    const int g    = lane >> 4;       // lane half (K group)
    const int ml   = lane & 15;
    const long bm0 = (long)blockIdx.y * GBM;
    const long bn0 = (long)blockIdx.x * GBN;

    v8f acc[2][4];
    const v8f vzero = {0.f,0.f,0.f,0.f,0.f,0.f,0.f,0.f};
#pragma unroll
    for (int i = 0; i < 2; ++i)
#pragma unroll
        for (int j = 0; j < 4; ++j) acc[i][j] = vzero;

    float4 ra[4], rb[4];

    auto issueLoads = [&](int kb) {
#pragma unroll
        for (int i = 0; i < 4; ++i) {
            int idx = tid + 256 * i;          // 0..1023 float4 slots
            int r   = idx >> 3;               // 0..127
            int c4  = idx & 7;                // 0..7
            ra[i] = *(const float4*)(A + (bm0 + r) * (long)K + (long)kb * GBK + c4 * 4);
            rb[i] = *(const float4*)(W + (bn0 + r) * (long)K + (long)kb * GBK + c4 * 4);
        }
    };
    auto cvtStore = [&](int buf) {
#pragma unroll
        for (int i = 0; i < 4; ++i) {
            int idx = tid + 256 * i;
            int r   = idx >> 3;
            int c4  = idx & 7;
            bf16x4 ta = {(bf16)ra[i].x, (bf16)ra[i].y, (bf16)ra[i].z, (bf16)ra[i].w};
            *(bf16x4*)&As[buf][r][c4 * 4] = ta;
            bf16x4 tb = {(bf16)rb[i].x, (bf16)rb[i].y, (bf16)rb[i].z, (bf16)rb[i].w};
            *(bf16x4*)&Bs[buf][r][c4 * 4] = tb;
        }
    };

    issueLoads(0);
    cvtStore(0);
    __syncthreads();

    const int nk = K / GBK;
    for (int kb = 0; kb < nk; ++kb) {
        const int cur = kb & 1;
        const bool more = (kb + 1 < nk);
        if (more) {
            issueLoads(kb + 1);               // loads in flight during WMMAs
            if (kb + 2 < nk) {                // stream f32 tiles into L2 early
                long pr = (tid < 128) ? (bm0 + tid) : (bn0 + tid - 128);
                const float* pp = (tid < 128)
                    ? (A + pr * (long)K + (long)(kb + 2) * GBK)
                    : (W + pr * (long)K + (long)(kb + 2) * GBK);
                __builtin_prefetch(pp, 0, 3);
            }
        }

        bf16x16 af[2], bfv[4];
#pragma unroll
        for (int mt = 0; mt < 2; ++mt) {
            // A-operand: lane=M row, 16 contiguous K at g*16
            const bf16* p = &As[cur][wm * 32 + mt * 16 + ml][g * 16];
            af[mt] = comb16(p, p + 8);
        }
#pragma unroll
        for (int nt = 0; nt < 4; ++nt) {
            // B-operand: lane=N col, 16 contiguous K at g*16
            const bf16* p = &Bs[cur][wn * 64 + nt * 16 + ml][g * 16];
            bfv[nt] = comb16(p, p + 8);
        }
#pragma unroll
        for (int mt = 0; mt < 2; ++mt)
#pragma unroll
            for (int nt = 0; nt < 4; ++nt)
                acc[mt][nt] = __builtin_amdgcn_wmma_f32_16x16x32_bf16(
                    false, af[mt], false, bfv[nt], (short)0, acc[mt][nt], false, false);

        if (more) cvtStore(cur ^ 1);          // waits for loads land HERE
        __syncthreads();
    }

    // Epilogue: C layout -> lane ml = N col, vgpr r = M row (+8 for hi half)
#pragma unroll
    for (int mt = 0; mt < 2; ++mt) {
#pragma unroll
        for (int nt = 0; nt < 4; ++nt) {
            const long colg = bn0 + wn * 64 + nt * 16 + ml;
            const float bv  = bias ? bias[colg] : 0.0f;
#pragma unroll
            for (int r = 0; r < 8; ++r) {
                const long rowg = bm0 + wm * 32 + mt * 16 + r + 8 * g;
                float v = acc[mt][nt][r] + bv;
                if (act == 2) {
                    // QKV split: col -> (which, head, d); row -> (b, t); write bf16
                    const int which = (int)(colg >> 10);
                    const int cc = (int)colg & 1023;
                    const int hh = cc >> 6, dd = cc & 63;
                    const int bb = (int)(rowg >> 10), tt = (int)rowg & 1023;
                    const size_t dst = (((size_t)(bb * 16 + hh)) * 1024 + tt) * 64 + dd;
                    const bf16 o = (bf16)v;
                    if (which == 0)      Qo[dst] = o;
                    else if (which == 1) Ko[dst] = o;
                    else                 Vo[dst] = o;
                } else {
                    if (res) v += res[rowg * (long)N + colg];
                    if (act == 1) v = 0.5f * v * (1.0f + erff(v * 0.70710678118f));
                    C[rowg * (long)N + colg] = v;
                }
            }
        }
    }
}

// ---------------------------------------------------------------------------
// Flash attention. Grid (T/64, H, B), 128 threads (4 waves).
// Wave w owns 16 query rows. Streams 64-key tiles; Q.K^T and P.V via WMMA.
// K tiles: double-buffered in LDS, filled by GLOBAL_LOAD_ASYNC_TO_LDS_B128
// (ASYNCcnt) when available -- no VGPR round trip, overlaps the WMMA block.
// V tiles: register-pipelined (transpose into LDS requires the VGPR path).
// ---------------------------------------------------------------------------
__global__ __launch_bounds__(128) void flash_attn_kernel(
    const bf16* __restrict__ Qb, const bf16* __restrict__ Kb,
    const bf16* __restrict__ Vb, float* __restrict__ O)
{
    const int qt   = blockIdx.x;   // 0..15
    const int h    = blockIdx.y;   // 0..15
    const int b    = blockIdx.z;   // 0..1
    const int tid  = threadIdx.x;
    const int lane = tid & 31;
    const int wave = tid >> 5;     // 0..3
    const int g    = lane >> 4;
    const int ml   = lane & 15;

    __shared__ __attribute__((aligned(16))) bf16 Qs[64][72];
    __shared__ __attribute__((aligned(16))) bf16 Ks[2][64][72];
    __shared__ __attribute__((aligned(16))) bf16 Vt[64][72];   // transposed: [d][key]
    __shared__ __attribute__((aligned(16))) bf16 Ps[4][16][72];

    const size_t head = (size_t)(b * 16 + h) * 1024 * 64;
    const bf16* Qh = Qb + head;
    const bf16* Kh = Kb + head;
    const bf16* Vh = Vb + head;

    // ---- Stage Q tile (64x64 bf16) ----
#if HAVE_ASYNC_LDS
#pragma unroll
    for (int i = 0; i < 4; ++i) {
        int idx = tid + 128 * i;       // 0..511, 8 elems each
        int r = idx >> 3, c = (idx & 7) * 8;
        __builtin_amdgcn_global_load_async_to_lds_b128(
            (as1_i32x4*)(void*)const_cast<bf16*>(Qh + (size_t)(qt * 64 + r) * 64 + c),
            (as3_i32x4*)(void*)&Qs[r][c], 0, 0);
    }
    __builtin_amdgcn_s_wait_asynccnt(0);
#else
#pragma unroll
    for (int i = 0; i < 4; ++i) {
        int idx = tid + 128 * i;
        int r = idx >> 3, c = (idx & 7) * 8;
        *(bf16x8*)&Qs[r][c] = *(const bf16x8*)(Qh + (size_t)(qt * 64 + r) * 64 + c);
    }
#endif
    __syncthreads();

    // Hoist Q fragments (invariant across key tiles)
    bf16x16 qf[2];
#pragma unroll
    for (int s = 0; s < 2; ++s) {
        const bf16* p = &Qs[wave * 16 + ml][s * 32 + g * 16];
        qf[s] = comb16(p, p + 8);
    }

    const v8f vzero = {0.f,0.f,0.f,0.f,0.f,0.f,0.f,0.f};
    v8f o[4];
#pragma unroll
    for (int c = 0; c < 4; ++c) o[c] = vzero;
    float mrow[8], lrow[8];
#pragma unroll
    for (int r = 0; r < 8; ++r) { mrow[r] = -1e30f; lrow[r] = 0.0f; }

    const float slope = -exp2f(-0.5f * (float)(h + 1));
    const float scale = 0.125f;                 // 1/sqrt(64)

    bf16x8 vreg[4];
#if !HAVE_ASYNC_LDS
    bf16x8 kreg[4];
#endif

    auto loadV = [&](int kt) {
#pragma unroll
        for (int i = 0; i < 4; ++i) {
            int idx = tid + 128 * i;
            int r = idx >> 3, c = (idx & 7) * 8;
            vreg[i] = *(const bf16x8*)(Vh + (size_t)(kt * 64 + r) * 64 + c);
        }
        if (kt + 1 < 16) {   // prefetch next tile's lines into L2
            __builtin_prefetch(Kh + (size_t)((kt + 1) * 64 + (tid >> 1)) * 64, 0, 3);
            __builtin_prefetch(Vh + (size_t)((kt + 1) * 64 + (tid >> 1)) * 64, 0, 3);
        }
    };
#if HAVE_ASYNC_LDS
    auto asyncK = [&](int kt, int buf) {
#pragma unroll
        for (int i = 0; i < 4; ++i) {
            int idx = tid + 128 * i;
            int r = idx >> 3, c = (idx & 7) * 8;
            __builtin_amdgcn_global_load_async_to_lds_b128(
                (as1_i32x4*)(void*)const_cast<bf16*>(Kh + (size_t)(kt * 64 + r) * 64 + c),
                (as3_i32x4*)(void*)&Ks[buf][r][c], 0, 0);
        }
    };
#else
    auto loadK = [&](int kt) {
#pragma unroll
        for (int i = 0; i < 4; ++i) {
            int idx = tid + 128 * i;
            int r = idx >> 3, c = (idx & 7) * 8;
            kreg[i] = *(const bf16x8*)(Kh + (size_t)(kt * 64 + r) * 64 + c);
        }
    };
    auto storeK = [&](int buf) {
#pragma unroll
        for (int i = 0; i < 4; ++i) {
            int idx = tid + 128 * i;
            int r = idx >> 3, c = (idx & 7) * 8;
            *(bf16x8*)&Ks[buf][r][c] = kreg[i];
        }
    };
#endif
    auto storeVt = [&]() {
#pragma unroll
        for (int i = 0; i < 4; ++i) {
            int idx = tid + 128 * i;
            int r = idx >> 3, c = (idx & 7) * 8;
#pragma unroll
            for (int e = 0; e < 8; ++e) Vt[c + e][r] = vreg[i][e];
        }
    };

    // Prologue: tile 0 in flight
#if HAVE_ASYNC_LDS
    asyncK(0, 0);
#else
    loadK(0);
#endif
    loadV(0);

    for (int kt = 0; kt < 16; ++kt) {
        const int cur = kt & 1;
        storeVt();                        // V loadcnt wait lands here
#if HAVE_ASYNC_LDS
        __builtin_amdgcn_s_wait_asynccnt(0);   // Ks[cur] resident
#else
        storeK(cur);
#endif
        __syncthreads();
        if (kt + 1 < 16) {
#if HAVE_ASYNC_LDS
            asyncK(kt + 1, cur ^ 1);      // fills other buffer during WMMAs
#else
            loadK(kt + 1);
#endif
            loadV(kt + 1);
        }

        // S = Q.K^T : 16(q) x 64(key) per wave
        v8f st[4];
#pragma unroll
        for (int nt = 0; nt < 4; ++nt) st[nt] = vzero;
#pragma unroll
        for (int s = 0; s < 2; ++s) {
#pragma unroll
            for (int nt = 0; nt < 4; ++nt) {
                const bf16* p = &Ks[cur][nt * 16 + ml][s * 32 + g * 16];
                bf16x16 kf = comb16(p, p + 8);
                st[nt] = __builtin_amdgcn_wmma_f32_16x16x32_bf16(
                    false, qf[s], false, kf, (short)0, st[nt], false, false);
            }
        }

        // Online softmax (row stats across the 16 lanes sharing a row)
#pragma unroll
        for (int r = 0; r < 8; ++r) {
            const int q = qt * 64 + wave * 16 + r + 8 * g;
            float sv[4];
            float tmax = -1e30f;
#pragma unroll
            for (int nt = 0; nt < 4; ++nt) {
                const int k = kt * 64 + nt * 16 + ml;
                float v = st[nt][r] * scale;
                if (q < 1023 && k < 1023) v += slope * fabsf((float)(q - k));
                sv[nt] = v;
                tmax = fmaxf(tmax, v);
            }
#pragma unroll
            for (int msk = 1; msk < 16; msk <<= 1)
                tmax = fmaxf(tmax, __shfl_xor(tmax, msk, 32));
            const float newm  = fmaxf(mrow[r], tmax);
            const float alpha = __expf(mrow[r] - newm);
            float psum = 0.0f;
#pragma unroll
            for (int nt = 0; nt < 4; ++nt) {
                float p = __expf(sv[nt] - newm);
                psum += p;
                Ps[wave][r + 8 * g][nt * 16 + ml] = (bf16)p;
            }
#pragma unroll
            for (int msk = 1; msk < 16; msk <<= 1)
                psum += __shfl_xor(psum, msk, 32);
            lrow[r] = lrow[r] * alpha + psum;
            mrow[r] = newm;
#pragma unroll
            for (int ct = 0; ct < 4; ++ct) o[ct][r] *= alpha;
        }

        // O += P.V  (A = P from per-wave LDS; B = V^T, K-contiguous)
#pragma unroll
        for (int s = 0; s < 2; ++s) {
            const bf16* pp = &Ps[wave][ml][s * 32 + g * 16];
            bf16x16 pf = comb16(pp, pp + 8);
#pragma unroll
            for (int ct = 0; ct < 4; ++ct) {
                const bf16* p = &Vt[ct * 16 + ml][s * 32 + g * 16];
                bf16x16 vf = comb16(p, p + 8);
                o[ct] = __builtin_amdgcn_wmma_f32_16x16x32_bf16(
                    false, pf, false, vf, (short)0, o[ct], false, false);
            }
        }
        __syncthreads();
    }

    // Write O in [B, T, H*DH] f32 for the out-projection GEMM
#pragma unroll
    for (int r = 0; r < 8; ++r) {
        const int q = qt * 64 + wave * 16 + r + 8 * g;
        const float inv = 1.0f / lrow[r];
        const size_t row = (size_t)(b * 1024 + q) * 1024 + h * 64;
#pragma unroll
        for (int ct = 0; ct < 4; ++ct)
            O[row + ct * 16 + ml] = o[ct][r] * inv;
    }
}

// ---------------------------------------------------------------------------
// LayerNorm over D=1024, one workgroup (256 thr) per row.
// ---------------------------------------------------------------------------
__global__ __launch_bounds__(256) void ln_kernel(
    const float* __restrict__ x, const float* __restrict__ gw,
    float* __restrict__ y)
{
    const int row = blockIdx.x;
    const int tid = threadIdx.x;
    const float* xr = x + (size_t)row * 1024;
    float4 v = *(const float4*)(xr + tid * 4);
    float s  = v.x + v.y + v.z + v.w;
    float s2 = v.x * v.x + v.y * v.y + v.z * v.z + v.w * v.w;
#pragma unroll
    for (int m = 1; m < 32; m <<= 1) {
        s  += __shfl_xor(s,  m, 32);
        s2 += __shfl_xor(s2, m, 32);
    }
    __shared__ float rs[8], rs2[8];
    const int wave = tid >> 5, lane = tid & 31;
    if (lane == 0) { rs[wave] = s; rs2[wave] = s2; }
    __syncthreads();
    float S = 0.f, S2 = 0.f;
#pragma unroll
    for (int i = 0; i < 8; ++i) { S += rs[i]; S2 += rs2[i]; }
    const float mu  = S * (1.0f / 1024.0f);
    const float var = S2 * (1.0f / 1024.0f) - mu * mu;
    const float rv  = rsqrtf(var + 1e-5f);
    const float* gp = gw + tid * 4;
    float4 ov;
    ov.x = (v.x - mu) * rv * gp[0];
    ov.y = (v.y - mu) * rv * gp[1];
    ov.z = (v.z - mu) * rv * gp[2];
    ov.w = (v.w - mu) * rv * gp[3];
    *(float4*)(y + (size_t)row * 1024 + tid * 4) = ov;
}

// ---------------------------------------------------------------------------
// Skip concat: XC[row, 0:1024] = X, XC[row, 1024:2048] = S * 2^-0.5
// ---------------------------------------------------------------------------
__global__ __launch_bounds__(256) void concat_scale_kernel(
    const float* __restrict__ x, const float* __restrict__ s,
    float* __restrict__ xc)
{
    const int idx = blockIdx.x * 256 + threadIdx.x;   // exact: 2048*2048
    const int row = idx >> 11;
    const int c   = idx & 2047;
    xc[idx] = (c < 1024) ? x[(size_t)row * 1024 + c]
                         : s[(size_t)row * 1024 + (c - 1024)] * 0.70710678118f;
}

// ---------------------------------------------------------------------------
// Host orchestration
// ---------------------------------------------------------------------------
extern "C" void kernel_launch(void* const* d_in, const int* in_sizes, int n_in,
                              void* d_out, int out_size, void* d_ws, size_t ws_size,
                              hipStream_t stream)
{
    const float* in_x       = (const float*)d_in[0];
    const float* attn_w     = (const float*)d_in[1];
    const float* attn_out_w = (const float*)d_in[2];
    const float* mlp_ln_g   = (const float*)d_in[3];
    const float* mlp_in_w   = (const float*)d_in[4];
    const float* mlp_in_b   = (const float*)d_in[5];
    const float* mlp_out_w  = (const float*)d_in[6];
    const float* mlp_out_b  = (const float*)d_in[7];
    const float* skip_w     = (const float*)d_in[8];
    const float* skip_b     = (const float*)d_in[9];
    const float* out_ln_g   = (const float*)d_in[10];

    const size_t MEG = 1u << 20;            // 1M floats
    float* ws   = (float*)d_ws;
    float* X    = ws;                        // [2048,1024]   2M
    float* XC   = ws + 2  * MEG;             // [2048,2048]   4M
    float* Obuf = ws + 8  * MEG;             // [2048,1024]   2M
    float* Hbuf = ws + 10 * MEG;             // [2048,1024]   2M
    float* FF   = ws + 12 * MEG;             // [2048,4096]   8M
    float* Ssav[3] = { ws + 20 * MEG, ws + 22 * MEG, ws + 24 * MEG }; // 3 x 2M
    bf16*  Qb   = (bf16*)(ws + 26 * MEG);    // [2,16,1024,64] bf16
    bf16*  Kb   = (bf16*)(ws + 27 * MEG);
    bf16*  Vb   = (bf16*)(ws + 28 * MEG);

    (void)hipMemcpyAsync(X, in_x, (size_t)2048 * 1024 * sizeof(float),
                         hipMemcpyDeviceToDevice, stream);

    for (int i = 0; i < 8; ++i) {
        if (i >= 5) {
            // s = connections.pop() : i=5 -> x4, i=6 -> x3, i=7 -> x2
            const float* Spop = (i == 5) ? Ssav[2] : (i == 6) ? Ssav[1] : Ssav[0];
            concat_scale_kernel<<<(2048 * 2048) / 256, 256, 0, stream>>>(X, Spop, XC);
            const int j = i - 4;
            gemm_bf16_kernel<<<dim3(1024 / 128, 2048 / 128), 256, 0, stream>>>(
                XC, skip_w + (size_t)j * 1024 * 2048, skip_b + (size_t)j * 1024,
                nullptr, X, nullptr, nullptr, nullptr, 2048, 1024, 2048, 0);
        }

        // ---- Attention ----
        // QKV projection with fused split + bf16 repack into [B,H,T,DH]
        gemm_bf16_kernel<<<dim3(3072 / 128, 2048 / 128), 256, 0, stream>>>(
            X, attn_w + (size_t)i * 3072 * 1024, nullptr, nullptr,
            nullptr, Qb, Kb, Vb, 2048, 3072, 1024, 2);
        flash_attn_kernel<<<dim3(16, 16, 2), 128, 0, stream>>>(Qb, Kb, Vb, Obuf);
        gemm_bf16_kernel<<<dim3(1024 / 128, 2048 / 128), 256, 0, stream>>>(
            Obuf, attn_out_w + (size_t)i * 1024 * 1024, nullptr, X,
            X, nullptr, nullptr, nullptr, 2048, 1024, 1024, 0);

        // Save connection after layers 2,3,4
        if (i >= 2 && i <= 4)
            (void)hipMemcpyAsync(Ssav[i - 2], X, (size_t)2048 * 1024 * sizeof(float),
                                 hipMemcpyDeviceToDevice, stream);

        // ---- MLP ----
        ln_kernel<<<2048, 256, 0, stream>>>(X, mlp_ln_g + (size_t)i * 1024, Hbuf);
        gemm_bf16_kernel<<<dim3(4096 / 128, 2048 / 128), 256, 0, stream>>>(
            Hbuf, mlp_in_w + (size_t)i * 4096 * 1024, mlp_in_b + (size_t)i * 4096,
            nullptr, FF, nullptr, nullptr, nullptr, 2048, 4096, 1024, 1 /*gelu*/);
        gemm_bf16_kernel<<<dim3(1024 / 128, 2048 / 128), 256, 0, stream>>>(
            FF, mlp_out_w + (size_t)i * 1024 * 4096, mlp_out_b + (size_t)i * 1024,
            X, X, nullptr, nullptr, nullptr, 2048, 1024, 4096, 0);
    }

    ln_kernel<<<2048, 256, 0, stream>>>(X, out_ln_g, (float*)d_out);
}
